// SelfAttention_67654324847607
// MI455X (gfx1250) — matrix operand order repaired and checked by reading, not compiled, not run on hardware
//
#include <hip/hip_runtime.h>
#include <hip/hip_bf16.h>

typedef __attribute__((ext_vector_type(16))) __bf16 v16bf;
typedef __attribute__((ext_vector_type(8)))  float  v8f;

#define WMMA_BF16(a, b, c) \
  __builtin_amdgcn_wmma_f32_16x16x32_bf16(false, (a), false, (b), (short)0, (c), false, false)

// Native bf16 converts (RNE) -- gfx1250 has hardware cvt, avoid manual bit math.
__device__ __forceinline__ unsigned pack_bf16x2(float a, float b) {
  unsigned short ua = __builtin_bit_cast(unsigned short, (__bf16)a);
  unsigned short ub = __builtin_bit_cast(unsigned short, (__bf16)b);
  return (unsigned)ua | ((unsigned)ub << 16);
}

// ---------------------------------------------------------------------------
// Kernel 1: qkv = x @ W_attn + b_attn   (M=8192, K=768, N=2304)
// Block: 256 thr (8 waves), macro-tile 64(M) x 64(N); each wave owns a 16x32
// strip -> 2 v_wmma per k-step with A-fragment reuse.
// Output scattered bf16: Q,K -> [B,H,T,Dh], V -> [B,H,Dh,T] (transposed).
// ---------------------------------------------------------------------------
__global__ __launch_bounds__(256)
void qkv_gemm_kernel(const float* __restrict__ X, const float* __restrict__ W,
                     const float* __restrict__ bias,
                     __bf16* __restrict__ Qb, __bf16* __restrict__ Kb,
                     __bf16* __restrict__ Vt)
{
  const int K = 768, N = 2304;
  __shared__ __bf16 As[64][36];   // [m][k], 72B pitch (b64/b32-store aligned, banks OK)
  __shared__ __bf16 Bs[64][36];   // [n][k] transposed
  const int tid = threadIdx.x, lane = tid & 31, wave = tid >> 5;
  const int waveM = wave & 3, waveN = wave >> 2;
  const int m0 = blockIdx.y * 64, n0 = blockIdx.x * 64;
  const int hi = lane >> 4, lo = lane & 15;

  v8f acc0 = {}, acc1 = {};
  for (int k0 = 0; k0 < K; k0 += 32) {
#pragma unroll
    for (int it = 0; it < 4; ++it) {        // A: 64x32 fp32, float2 loads, packed b32 stores
      int f = it * 256 + tid;
      int m = f >> 4, kp = (f & 15) * 2;
      float2 v = *reinterpret_cast<const float2*>(&X[(size_t)(m0 + m) * K + k0 + kp]);
      *reinterpret_cast<unsigned*>(&As[m][kp]) = pack_bf16x2(v.x, v.y);
    }
#pragma unroll
    for (int it = 0; it < 4; ++it) {        // B: 32x64 (k x n) -> Bs[n][k]
      int f = it * 256 + tid;
      int k = f >> 5, n = (f & 31) * 2;
      float2 v = *reinterpret_cast<const float2*>(&W[(size_t)(k0 + k) * N + n0 + n]);
      Bs[n][k]     = (__bf16)v.x;
      Bs[n + 1][k] = (__bf16)v.y;
    }
    __syncthreads();
    v16bf a, b0, b1;
#pragma unroll
    for (int j = 0; j < 8; ++j) {
      int ka = (j < 4) ? (hi * 8 + 2 * j) : (16 + hi * 8 + 2 * (j - 4));
      a[2 * j]      = As[waveM * 16 + lo][ka];
      a[2 * j + 1]  = As[waveM * 16 + lo][ka + 1];
      int kb = (j < 4) ? (hi * 8 + 2 * j) : (16 + hi * 8 + 2 * (j - 4));
      b0[2 * j]     = Bs[waveN * 32 + lo][kb];
      b0[2 * j + 1] = Bs[waveN * 32 + lo][kb + 1];
      b1[2 * j]     = Bs[waveN * 32 + 16 + lo][kb];
      b1[2 * j + 1] = Bs[waveN * 32 + 16 + lo][kb + 1];
    }
    acc0 = WMMA_BF16(a, b0, acc0);
    acc1 = WMMA_BF16(a, b1, acc1);
    __syncthreads();
  }

#pragma unroll
  for (int nt = 0; nt < 2; ++nt) {
    const v8f& acc = nt ? acc1 : acc0;
    const int gn = n0 + waveN * 32 + nt * 16 + lo;
    const float bv  = bias[gn];
    const int which = gn / 768;
    const int rem   = gn % 768;
    const int hd = rem >> 6, d = rem & 63;
#pragma unroll
    for (int r = 0; r < 8; ++r) {
      int gm = m0 + waveM * 16 + hi * 8 + r;  // row = b*T + t
      int bb = gm >> 10, t = gm & 1023;
      __bf16 v = (__bf16)(acc[r] + bv);
      if (which == 0)      Qb[((size_t)(bb * 12 + hd) * 1024 + t) * 64 + d] = v;
      else if (which == 1) Kb[((size_t)(bb * 12 + hd) * 1024 + t) * 64 + d] = v;
      else                 Vt[((size_t)(bb * 12 + hd) * 64 + d) * 1024 + t] = v;
    }
  }
}

// ---------------------------------------------------------------------------
// Kernel 2: causal flash attention. One (b,h) + 128 query rows per block,
// one wave per 16 query rows, divergent causal key loop (no barriers).
// K tiles are staged through LDS via GLOBAL_LOAD_ASYNC_TO_LDS_B128 (ASYNCcnt,
// inline asm -- the clang builtin isn't portable across toolchains), with a
// 136B row pitch so the 16-row fragment gather is bank-conflict free.
// 4 v_wmma (QK^T) + online softmax + 4 v_wmma (P@V) per 32-key block.
// ---------------------------------------------------------------------------
__global__ __launch_bounds__(256)
void attn_kernel(const __bf16* __restrict__ Qb, const __bf16* __restrict__ Kb,
                 const __bf16* __restrict__ Vt, __bf16* __restrict__ Y)
{
  __shared__ __bf16 Kt[8][32][68];   // per-wave K tile [key][ch], 136B pitch
  __shared__ __bf16 Pl[8][16][36];   // per-wave P tile [row][key]
  const int tid = threadIdx.x, lane = tid & 31, wave = tid >> 5;
  const int hi = lane >> 4, lo = lane & 15;
  const int bh = blockIdx.y;                       // b*12 + h
  const int b  = bh / 12, h = bh % 12;
  const int q0 = blockIdx.x * 128 + wave * 16;
  const __bf16* Qp = Qb + (size_t)bh * 1024 * 64;  // [t][d]
  const __bf16* Kp = Kb + (size_t)bh * 1024 * 64;  // [t][d]
  const __bf16* Vp = Vt + (size_t)bh * 64 * 1024;  // [d][t]

  const unsigned ktLds = (unsigned)(unsigned long long)&Kt[wave][0][0];
  const int row4 = lane >> 3, chunk = lane & 7;    // 16B chunks: 8 per 128B key row

  // Q A-fragments (16 rows x 64 ch), resident in VGPRs for the whole loop
  v16bf aq[2];
#pragma unroll
  for (int half = 0; half < 2; ++half) {
#pragma unroll
    for (int j = 0; j < 8; ++j) {
      int ka = ((j < 4) ? (hi * 8 + 2 * j) : (16 + hi * 8 + 2 * (j - 4))) + half * 32;
      const __bf16* p = Qp + (size_t)(q0 + lo) * 64 + ka;
      aq[half][2 * j]     = p[0];
      aq[half][2 * j + 1] = p[1];
    }
  }

  v8f o0 = {}, o1 = {}, o2 = {}, o3 = {};
  float mR[8], lR[8];
#pragma unroll
  for (int r = 0; r < 8; ++r) { mR[r] = -1e30f; lR[r] = 0.0f; }

  const int kend = q0 + 15;
  for (int kb = 0; kb * 32 <= kend; ++kb) {
    const int kbase = kb * 32;

    // ---- async-stage K tile (32 keys x 64 ch bf16 = 4KB) into this wave's LDS ----
    asm volatile("s_wait_dscnt 0" ::: "memory");   // WAR vs previous tile's ds reads
#pragma unroll
    for (int t = 0; t < 8; ++t) {
      unsigned loff = ktLds + (unsigned)((t * 4 + row4) * 136 + chunk * 16);
      unsigned goff = (unsigned)(((kbase + t * 4 + row4) << 7) + chunk * 16);
      asm volatile("global_load_async_to_lds_b128 %0, %1, %2"
                   :: "v"(loff), "v"(goff), "s"(Kp) : "memory");
    }
    asm volatile("s_wait_asynccnt 0" ::: "memory");

    // ---- S = Q K^T for 32 keys (two 16-key C tiles) ----
    v8f s0 = {}, s1 = {};
#pragma unroll
    for (int half = 0; half < 2; ++half) {
      v16bf bk0, bk1;
#pragma unroll
      for (int j = 0; j < 8; ++j) {
        int kc = ((j < 4) ? (hi * 8 + 2 * j) : (16 + hi * 8 + 2 * (j - 4))) + half * 32;      // channel
        bk0[2 * j]     = Kt[wave][lo][kc];
        bk0[2 * j + 1] = Kt[wave][lo][kc + 1];
        bk1[2 * j]     = Kt[wave][16 + lo][kc];
        bk1[2 * j + 1] = Kt[wave][16 + lo][kc + 1];
      }
      s0 = WMMA_BF16(aq[half], bk0, s0);
      s1 = WMMA_BF16(aq[half], bk1, s1);
    }

    // ---- online softmax over this 32-key block ----
#pragma unroll
    for (int r = 0; r < 8; ++r) {
      const int qrow = q0 + hi * 8 + r;
      const int c0 = kbase + lo, c1 = kbase + 16 + lo;
      float f0 = (c0 <= qrow) ? s0[r] * 0.125f : -1e30f;   // 1/sqrt(64)
      float f1 = (c1 <= qrow) ? s1[r] * 0.125f : -1e30f;
      float rm = fmaxf(f0, f1);
#pragma unroll
      for (int dx = 1; dx < 16; dx <<= 1) rm = fmaxf(rm, __shfl_xor(rm, dx, 32));
      float mnew  = fmaxf(mR[r], rm);
      float alpha = __expf(mR[r] - mnew);
      float p0 = __expf(f0 - mnew);
      float p1 = __expf(f1 - mnew);
      float rs = p0 + p1;
#pragma unroll
      for (int dx = 1; dx < 16; dx <<= 1) rs += __shfl_xor(rs, dx, 32);
      lR[r] = lR[r] * alpha + rs;
      mR[r] = mnew;
      const int row = hi * 8 + r;
      Pl[wave][row][lo]      = (__bf16)p0;
      Pl[wave][row][16 + lo] = (__bf16)p1;
      o0[r] *= alpha; o1[r] *= alpha; o2[r] *= alpha; o3[r] *= alpha;
    }

    // ---- O += P(16x32) @ V(32x64); V read transposed [d][t], pairs contiguous ----
    v16bf ap;
#pragma unroll
    for (int j = 0; j < 8; ++j) {
      int ka = (j < 4) ? (hi * 8 + 2 * j) : (16 + hi * 8 + 2 * (j - 4));
      ap[2 * j]     = Pl[wave][lo][ka];
      ap[2 * j + 1] = Pl[wave][lo][ka + 1];
    }
#pragma unroll
    for (int nt = 0; nt < 4; ++nt) {
      v16bf bvv;
#pragma unroll
      for (int j = 0; j < 8; ++j) {
        int key = kbase + ((j < 4) ? (hi * 8 + 2 * j) : (16 + hi * 8 + 2 * (j - 4)));
        const __bf16* p = Vp + (size_t)(nt * 16 + lo) * 1024 + key;
        bvv[2 * j] = p[0]; bvv[2 * j + 1] = p[1];
      }
      if      (nt == 0) o0 = WMMA_BF16(ap, bvv, o0);
      else if (nt == 1) o1 = WMMA_BF16(ap, bvv, o1);
      else if (nt == 2) o2 = WMMA_BF16(ap, bvv, o2);
      else              o3 = WMMA_BF16(ap, bvv, o3);
    }
  }

  // ---- normalize + write Y[b][t][h*64 + c] (bf16) ----
#pragma unroll
  for (int r = 0; r < 8; ++r) {
    const int t = q0 + hi * 8 + r;
    const float inv = 1.0f / lR[r];
    size_t base = ((size_t)b * 1024 + t) * 768 + h * 64;
    Y[base +  0 + lo] = (__bf16)(o0[r] * inv);
    Y[base + 16 + lo] = (__bf16)(o1[r] * inv);
    Y[base + 32 + lo] = (__bf16)(o2[r] * inv);
    Y[base + 48 + lo] = (__bf16)(o3[r] * inv);
  }
}

// ---------------------------------------------------------------------------
// Kernel 3: out = Y @ W_proj + b_proj   (M=8192, K=768, N=768), fp32 out.
// Same 64x64 macro-tile / 16x32-per-wave structure; A is already bf16.
// ---------------------------------------------------------------------------
__global__ __launch_bounds__(256)
void proj_gemm_kernel(const __bf16* __restrict__ Yb, const float* __restrict__ W,
                      const float* __restrict__ bias, float* __restrict__ Out)
{
  const int K = 768, N = 768;
  __shared__ __bf16 As[64][36];
  __shared__ __bf16 Bs[64][36];
  const int tid = threadIdx.x, lane = tid & 31, wave = tid >> 5;
  const int waveM = wave & 3, waveN = wave >> 2;
  const int m0 = blockIdx.y * 64, n0 = blockIdx.x * 64;
  const int hi = lane >> 4, lo = lane & 15;

  v8f acc0 = {}, acc1 = {};
  for (int k0 = 0; k0 < K; k0 += 32) {
#pragma unroll
    for (int it = 0; it < 4; ++it) {        // A: bf16, b32 copy
      int f = it * 256 + tid;
      int m = f >> 4, kp = (f & 15) * 2;
      *reinterpret_cast<unsigned*>(&As[m][kp]) =
          *reinterpret_cast<const unsigned*>(&Yb[(size_t)(m0 + m) * K + k0 + kp]);
    }
#pragma unroll
    for (int it = 0; it < 4; ++it) {        // B: fp32 -> bf16, transposed
      int f = it * 256 + tid;
      int k = f >> 5, n = (f & 31) * 2;
      float2 v = *reinterpret_cast<const float2*>(&W[(size_t)(k0 + k) * N + n0 + n]);
      Bs[n][k]     = (__bf16)v.x;
      Bs[n + 1][k] = (__bf16)v.y;
    }
    __syncthreads();
    v16bf a, b0, b1;
#pragma unroll
    for (int j = 0; j < 8; ++j) {
      int ka = (j < 4) ? (hi * 8 + 2 * j) : (16 + hi * 8 + 2 * (j - 4));
      a[2 * j]      = As[waveM * 16 + lo][ka];
      a[2 * j + 1]  = As[waveM * 16 + lo][ka + 1];
      int kb = (j < 4) ? (hi * 8 + 2 * j) : (16 + hi * 8 + 2 * (j - 4));
      b0[2 * j]     = Bs[waveN * 32 + lo][kb];
      b0[2 * j + 1] = Bs[waveN * 32 + lo][kb + 1];
      b1[2 * j]     = Bs[waveN * 32 + 16 + lo][kb];
      b1[2 * j + 1] = Bs[waveN * 32 + 16 + lo][kb + 1];
    }
    acc0 = WMMA_BF16(a, b0, acc0);
    acc1 = WMMA_BF16(a, b1, acc1);
    __syncthreads();
  }

#pragma unroll
  for (int nt = 0; nt < 2; ++nt) {
    const v8f& acc = nt ? acc1 : acc0;
    const int gn = n0 + waveN * 32 + nt * 16 + lo;
    const float bv = bias[gn];
#pragma unroll
    for (int r = 0; r < 8; ++r) {
      int gm = m0 + waveM * 16 + hi * 8 + r;
      Out[(size_t)gm * N + gn] = acc[r] + bv;
    }
  }
}

// ---------------------------------------------------------------------------
extern "C" void kernel_launch(void* const* d_in, const int* in_sizes, int n_in,
                              void* d_out, int out_size, void* d_ws, size_t ws_size,
                              hipStream_t stream) {
  const float* x      = (const float*)d_in[0];   // [8,1024,768]
  const float* W_attn = (const float*)d_in[1];   // [768,2304]
  const float* b_attn = (const float*)d_in[2];   // [2304]
  const float* W_proj = (const float*)d_in[3];   // [768,768]
  const float* b_proj = (const float*)d_in[4];   // [768]
  float* out = (float*)d_out;                    // [8,1024,768] fp32

  // Workspace (bf16): Q [B,H,T,Dh] | K [B,H,T,Dh] | V^T [B,H,Dh,T] | Y [B,T,C]
  char* ws = (char*)d_ws;
  const size_t per = (size_t)8 * 12 * 1024 * 64 * sizeof(__bf16);
  __bf16* Qb = (__bf16*)(ws);
  __bf16* Kb = (__bf16*)(ws + per);
  __bf16* Vt = (__bf16*)(ws + 2 * per);
  __bf16* Yb = (__bf16*)(ws + 3 * per);

  qkv_gemm_kernel<<<dim3(36, 128), dim3(256), 0, stream>>>(x, W_attn, b_attn, Qb, Kb, Vt);
  attn_kernel<<<dim3(8, 96), dim3(256), 0, stream>>>(Qb, Kb, Vt, Yb);
  proj_gemm_kernel<<<dim3(12, 128), dim3(256), 0, stream>>>(Yb, W_proj, b_proj, out);
}
